// ModuleSelfAttention_9491877724819
// MI455X (gfx1250) — matrix operand order repaired and checked by reading, not compiled, not run on hardware
//
#include <hip/hip_runtime.h>

// ---------------------------------------------------------------------------
// Self-attention for MI455X (gfx1250): all GEMMs on v_wmma_f32_16x16x32_bf16,
// f32 softmax/epilogue. Problem constants from the reference.
// Round-2 change: re-tiled k_attn (16 waves x 8 tiles) and k_out (8 tiles/wave)
// to eliminate the VGPR spills seen in the round-1 assembly.
// ---------------------------------------------------------------------------

typedef __bf16 bf16_t;
typedef __attribute__((ext_vector_type(16))) __bf16 v16bf;
typedef __attribute__((ext_vector_type(8)))  __bf16 v8bf;
typedef __attribute__((ext_vector_type(8)))  float  v8f;

namespace {
constexpr int kB  = 16;
constexpr int kC  = 1024;
constexpr int kT  = 2048;
constexpr int kDK = 128;
}

// ----- WMMA fragment helpers (contiguous K order, same for A and B) -----
// A (16x32 bf16): lane holds row M = lane&15; contiguous K-chunk
// [16h, 16h+16) where h = lane>>4. Two 16-byte loads.
__device__ __forceinline__ v16bf load_frag_a(const bf16_t* __restrict__ base,
                                             int lda, int mbase, int k0, int lane) {
  const bf16_t* p = base + (size_t)(mbase + (lane & 15)) * lda
                         + (size_t)(k0 + ((lane >> 4) << 4));
  union { v16bf v; v8bf h[2]; } u;
  u.h[0] = *(const v8bf*)(p);
  u.h[1] = *(const v8bf*)(p + 8);
  return u.v;
}

// B (32x16 bf16), stored as Bt row-major (N x K): lane holds column
// N = lane&15, contiguous K-chunk [16h, 16h+16). One 32-byte load.
__device__ __forceinline__ v16bf load_frag_b(const bf16_t* __restrict__ base,
                                             int ldb, int nbase, int k0, int lane) {
  const bf16_t* p = base + (size_t)(nbase + (lane & 15)) * ldb
                         + (size_t)(k0 + ((lane >> 4) << 4));
  return *(const v16bf*)(p);
}

__device__ __forceinline__ v8f wmma_bf16(v16bf a, v16bf b, v8f c) {
  // (neg_a, A, neg_b, B, c_mod, C, reuse_a, reuse_b)
  return __builtin_amdgcn_wmma_f32_16x16x32_bf16(false, a, false, b,
                                                 (short)0, c, false, false);
}

// ---------------------------------------------------------------------------
// Stage 0a: x (B,C,T) f32 -> xT (B,T,C) bf16, LDS-tiled transpose.
// ---------------------------------------------------------------------------
__global__ void k_transpose_x(const float* __restrict__ x, bf16_t* __restrict__ xT) {
  __shared__ float tile[32][33];
  const int b  = blockIdx.z;
  const int t0 = blockIdx.x * 32;
  const int c0 = blockIdx.y * 32;
  const int tx = threadIdx.x;   // 0..31
  const int ty = threadIdx.y;   // 0..7
  const float* xb = x + (size_t)b * kC * kT;
  #pragma unroll
  for (int i = 0; i < 32; i += 8)
    tile[ty + i][tx] = xb[(size_t)(c0 + ty + i) * kT + (t0 + tx)];
  __syncthreads();
  bf16_t* xTb = xT + (size_t)b * kT * kC;
  #pragma unroll
  for (int i = 0; i < 32; i += 8)
    xTb[(size_t)(t0 + ty + i) * kC + (c0 + tx)] = (bf16_t)tile[tx][ty + i];
}

// Stage 0b: weight f32 -> bf16 (no transpose: W is (O,C) == Bt (N,K)).
__global__ void k_f32_to_bf16(const float* __restrict__ in, bf16_t* __restrict__ out, int n) {
  int i = blockIdx.x * blockDim.x + threadIdx.x;
  if (i < n) out[i] = (bf16_t)in[i];
}

// ---------------------------------------------------------------------------
// Stage 1: projection GEMM. out(t,o) = sum_c xT(t,c)*W(o,c) + bias(o).
// A = xT (T x C), Bt = W (N x C). 4 waves/block, 2 N-tiles per wave.
// TRANSPOSED=false: store (T,N) bf16 (for q, kT).
// TRANSPOSED=true : store (N,T) bf16 (for v, so the out-GEMM A is contiguous).
// ---------------------------------------------------------------------------
template <bool TRANSPOSED>
__global__ __launch_bounds__(128)
void k_proj(const bf16_t* __restrict__ xT, const bf16_t* __restrict__ W,
            const float* __restrict__ bias, bf16_t* __restrict__ out, int N) {
  const int b     = blockIdx.z;
  const int mbase = blockIdx.x * 16;
  const int lane  = threadIdx.x & 31;
  const int wave  = threadIdx.x >> 5;
  const int nbase = (blockIdx.y * 4 + wave) * 32;  // two 16-wide tiles
  const bf16_t* A = xT + (size_t)b * kT * kC;

  v8f acc0 = {}; v8f acc1 = {};
  for (int k0 = 0; k0 < kC; k0 += 32) {
    v16bf a  = load_frag_a(A, kC, mbase, k0, lane);
    v16bf b0 = load_frag_b(W, kC, nbase,      k0, lane);
    v16bf b1 = load_frag_b(W, kC, nbase + 16, k0, lane);
    acc0 = wmma_bf16(a, b0, acc0);
    acc1 = wmma_bf16(a, b1, acc1);
  }

  const int col0 = nbase + (lane & 15);
  const int rowh = mbase + 8 * (lane >> 4);
  const float bia0 = bias[col0];
  const float bia1 = bias[col0 + 16];
  if (!TRANSPOSED) {
    bf16_t* ob = out + (size_t)b * kT * N;
    #pragma unroll
    for (int r = 0; r < 8; ++r) {
      ob[(size_t)(rowh + r) * N + col0]      = (bf16_t)(acc0[r] + bia0);
      ob[(size_t)(rowh + r) * N + col0 + 16] = (bf16_t)(acc1[r] + bia1);
    }
  } else {
    bf16_t* ob = out + (size_t)b * (size_t)N * kT;  // (N, T)
    #pragma unroll
    for (int r = 0; r < 8; ++r) {
      ob[(size_t)col0 * kT + rowh + r]        = (bf16_t)(acc0[r] + bia0);
      ob[(size_t)(col0 + 16) * kT + rowh + r] = (bf16_t)(acc1[r] + bia1);
    }
  }
}

// ---------------------------------------------------------------------------
// Stage 2: energy + softmax. One workgroup = 16 query rows x all 2048 keys.
// 16 waves x 8 N-tiles each (64 acc VGPRs/wave -> no spills), K = 128.
// Emits unnormalized P = exp(e - rowmax) as bf16 and rowscale = 1/rowsum.
// ---------------------------------------------------------------------------
__global__ __launch_bounds__(512)
void k_attn(const bf16_t* __restrict__ q, const bf16_t* __restrict__ kTm,
            bf16_t* __restrict__ P, float* __restrict__ rowscale) {
  const int b     = blockIdx.z;
  const int mbase = blockIdx.x * 16;
  const int lane  = threadIdx.x & 31;
  const int wave  = threadIdx.x >> 5;          // 0..15
  const bf16_t* qb = q   + (size_t)b * kT * kDK;
  const bf16_t* kb = kTm + (size_t)b * kT * kDK;

  v16bf afr[4];
  #pragma unroll
  for (int kk = 0; kk < 4; ++kk) afr[kk] = load_frag_a(qb, kDK, mbase, kk * 32, lane);

  v8f acc[8];
  const v8f vzero = {};
  #pragma unroll
  for (int t = 0; t < 8; ++t) acc[t] = vzero;

  const int ncol0 = wave * 128;                // 8 tiles of 16 columns
  for (int t = 0; t < 8; ++t) {
    const int nb = ncol0 + t * 16;
    #pragma unroll
    for (int kk = 0; kk < 4; ++kk) {
      v16bf bf = load_frag_b(kb, kDK, nb, kk * 32, lane);
      acc[t] = wmma_bf16(afr[kk], bf, acc[t]);
    }
  }

  __shared__ float red[16][16];
  __shared__ float gmax[16];

  // per-lane row max over this wave's 128 columns, then half-wave reduce
  float lm[8];
  #pragma unroll
  for (int r = 0; r < 8; ++r) {
    float m = acc[0][r];
    #pragma unroll
    for (int t = 1; t < 8; ++t) m = fmaxf(m, acc[t][r]);
    for (int off = 1; off < 16; off <<= 1) m = fmaxf(m, __shfl_xor(m, off, 32));
    lm[r] = m;
  }
  if ((lane & 15) == 0) {
    const int rb = (lane >> 4) * 8;
    #pragma unroll
    for (int r = 0; r < 8; ++r) red[wave][rb + r] = lm[r];
  }
  __syncthreads();
  if (threadIdx.x < 16) {
    float m = red[0][threadIdx.x];
    #pragma unroll
    for (int w = 1; w < 16; ++w) m = fmaxf(m, red[w][threadIdx.x]);
    gmax[threadIdx.x] = m;
  }
  __syncthreads();

  // exp, bf16-round in place, row sums
  float ls[8];
  #pragma unroll
  for (int r = 0; r < 8; ++r) {
    const float m = gmax[r + 8 * (lane >> 4)];
    float s = 0.f;
    #pragma unroll
    for (int t = 0; t < 8; ++t) {
      float p = __expf(acc[t][r] - m);
      bf16_t pb = (bf16_t)p;        // round now so sum matches stored P
      float pf = (float)pb;
      acc[t][r] = pf;
      s += pf;
    }
    for (int off = 1; off < 16; off <<= 1) s += __shfl_xor(s, off, 32);
    ls[r] = s;
  }
  if ((lane & 15) == 0) {
    const int rb = (lane >> 4) * 8;
    #pragma unroll
    for (int r = 0; r < 8; ++r) red[wave][rb + r] = ls[r];
  }
  __syncthreads();
  if (threadIdx.x < 16) {
    float s = 0.f;
    #pragma unroll
    for (int w = 0; w < 16; ++w) s += red[w][threadIdx.x];
    rowscale[(size_t)b * kT + mbase + threadIdx.x] = 1.0f / s;
  }

  // store unnormalized P (bf16)
  bf16_t* Pb = P + (size_t)b * kT * kT + (size_t)mbase * kT;
  const int coll = lane & 15;
  const int rowh = 8 * (lane >> 4);
  for (int t = 0; t < 8; ++t) {
    const int nb = ncol0 + t * 16;
    #pragma unroll
    for (int r = 0; r < 8; ++r)
      Pb[(size_t)(rowh + r) * kT + nb + coll] = (bf16_t)acc[t][r];
  }
}

// ---------------------------------------------------------------------------
// Stage 3: out(c,i) = gamma * rowscale(i) * sum_j v(c,j) P(i,j) + x(c,i).
// A = v (C x T), Bt = P (T x T). 4 waves/block, 8 N-tiles per wave
// (64 acc VGPRs/wave; A fragment amortized over 8 WMMAs).
// ---------------------------------------------------------------------------
__global__ __launch_bounds__(128)
void k_out(const bf16_t* __restrict__ v, const bf16_t* __restrict__ P,
           const float* __restrict__ rowscale, const float* __restrict__ x,
           const float* __restrict__ gamma, float* __restrict__ out) {
  const int b     = blockIdx.z;
  const int mbase = blockIdx.x * 16;                // channel c
  const int lane  = threadIdx.x & 31;
  const int wave  = threadIdx.x >> 5;
  const int nbase = (blockIdx.y * 4 + wave) * 128;  // position i, 8 tiles
  const bf16_t* vb = v + (size_t)b * kC * kT;
  const bf16_t* Pb = P + (size_t)b * kT * kT;

  v8f acc[8];
  const v8f vzero = {};
  #pragma unroll
  for (int t = 0; t < 8; ++t) acc[t] = vzero;

  for (int k0 = 0; k0 < kT; k0 += 32) {
    if (k0 + 32 < kT)  // prefetch next A chunk -> global_prefetch_b8
      __builtin_prefetch(vb + (size_t)(mbase + (lane & 15)) * kT + k0 + 32, 0, 1);
    v16bf a = load_frag_a(vb, kT, mbase, k0, lane);
    #pragma unroll
    for (int t = 0; t < 8; ++t) {
      v16bf bb = load_frag_b(Pb, kT, nbase + t * 16, k0, lane);
      acc[t] = wmma_bf16(a, bb, acc[t]);
    }
  }

  const float g = gamma[0];
  const int coll = lane & 15;
  const int rowh = mbase + 8 * (lane >> 4);
  const float* xb = x   + (size_t)b * kC * kT;
  float*       ob = out + (size_t)b * kC * kT;
  const float* rs = rowscale + (size_t)b * kT;
  #pragma unroll
  for (int t = 0; t < 8; ++t) {
    const int i = nbase + t * 16 + coll;
    const float sc = g * rs[i];
    #pragma unroll
    for (int r = 0; r < 8; ++r) {
      const size_t idx = (size_t)(rowh + r) * kT + i;
      ob[idx] = sc * acc[t][r] + xb[idx];
    }
  }
}

// ---------------------------------------------------------------------------
extern "C" void kernel_launch(void* const* d_in, const int* in_sizes, int n_in,
                              void* d_out, int out_size, void* d_ws, size_t ws_size,
                              hipStream_t stream) {
  (void)in_sizes; (void)n_in; (void)out_size; (void)ws_size;
  const float* x     = (const float*)d_in[0];
  const float* Wq    = (const float*)d_in[1];
  const float* bq    = (const float*)d_in[2];
  const float* Wk    = (const float*)d_in[3];
  const float* bk    = (const float*)d_in[4];
  const float* Wv    = (const float*)d_in[5];
  const float* bv    = (const float*)d_in[6];
  const float* gamma = (const float*)d_in[7];
  float* out = (float*)d_out;

  // Workspace carve-up (256 B aligned): ~275 MiB total.
  char* ws = (char*)d_ws;
  size_t off = 0;
  auto carve = [&](size_t bytes) {
    void* p = ws + off;
    off += (bytes + 255) & ~(size_t)255;
    return p;
  };
  bf16_t* xT   = (bf16_t*)carve((size_t)kB * kT * kC  * sizeof(bf16_t));  //  64 MiB
  bf16_t* Wqb  = (bf16_t*)carve((size_t)kDK * kC      * sizeof(bf16_t));
  bf16_t* Wkb  = (bf16_t*)carve((size_t)kDK * kC      * sizeof(bf16_t));
  bf16_t* Wvb  = (bf16_t*)carve((size_t)kC * kC       * sizeof(bf16_t));  //   2 MiB
  bf16_t* qw   = (bf16_t*)carve((size_t)kB * kT * kDK * sizeof(bf16_t));  //   8 MiB
  bf16_t* kTw  = (bf16_t*)carve((size_t)kB * kT * kDK * sizeof(bf16_t));  //   8 MiB
  bf16_t* vw   = (bf16_t*)carve((size_t)kB * kC * kT  * sizeof(bf16_t));  //  64 MiB
  bf16_t* Pw   = (bf16_t*)carve((size_t)kB * kT * kT  * sizeof(bf16_t));  // 128 MiB
  float*  rsw  = (float*) carve((size_t)kB * kT       * sizeof(float));

  // Stage 0: transpose/convert activations + weights.
  k_transpose_x<<<dim3(kT / 32, kC / 32, kB), dim3(32, 8), 0, stream>>>(x, xT);
  {
    int n = kDK * kC;
    k_f32_to_bf16<<<(n + 255) / 256, 256, 0, stream>>>(Wq, Wqb, n);
    k_f32_to_bf16<<<(n + 255) / 256, 256, 0, stream>>>(Wk, Wkb, n);
    n = kC * kC;
    k_f32_to_bf16<<<(n + 255) / 256, 256, 0, stream>>>(Wv, Wvb, n);
  }

  // Stage 1: projections. WG covers 16 rows x 128 cols.
  k_proj<false><<<dim3(kT / 16, kDK / 128, kB), 128, 0, stream>>>(xT, Wqb, bq, qw, kDK);
  k_proj<false><<<dim3(kT / 16, kDK / 128, kB), 128, 0, stream>>>(xT, Wkb, bk, kTw, kDK);
  k_proj<true ><<<dim3(kT / 16, kC  / 128, kB), 128, 0, stream>>>(xT, Wvb, bv, vw, kC);

  // Stage 2: energy + softmax (16 rows x full 2048 keys per WG, 16 waves).
  k_attn<<<dim3(kT / 16, 1, kB), 512, 0, stream>>>(qw, kTw, Pw, rsw);

  // Stage 3: attention-weighted values + residual epilogue.
  k_out<<<dim3(kC / 16, kT / 512, kB), 128, 0, stream>>>(vw, Pw, rsw, x, gamma, out);
}
